// Attn_52432960749709
// MI455X (gfx1250) — matrix operand order repaired and checked by reading, not compiled, not run on hardware
//
#include <hip/hip_runtime.h>
#include <stdint.h>

#define B_ 8
#define N_ 2048
#define D_ 512

typedef __bf16 bf16_t;
typedef __attribute__((ext_vector_type(16))) __bf16 v16bf;
typedef __attribute__((ext_vector_type(8)))  float  v8f;

// ---------- helpers ----------

__device__ __forceinline__ uint16_t f32_to_bf16_bits(float x) {
  union { float f; uint32_t u; } c; c.f = x;
  uint32_t r = c.u + 0x7FFFu + ((c.u >> 16) & 1u);   // round-to-nearest-even
  return (uint16_t)(r >> 16);
}

// A fragment: 16x32 bf16, row-major source, lda in elements.
// ISA layout: lanes 0-15 -> M=lane, K = {0..7, 16..23}; lanes 16-31 -> M=lane-16, K = {8..15, 24..31}
__device__ __forceinline__ v16bf load_a_frag(const bf16_t* A, int lda, int row0, int k0, int lane) {
  const bf16_t* p = A + (size_t)(row0 + (lane & 15)) * lda + k0 + ((lane >> 4) << 3);
  union { uint4 u[2]; v16bf v; } t;
  t.u[0] = *(const uint4*)(p);        // K +0..7   (16B)
  t.u[1] = *(const uint4*)(p + 16);   // K +16..23 (16B)
  return t.v;
}

// B fragment: 32x16 bf16 supplied from BT stored row-major [col][k], ldb in elements.
// ISA layout: lanes 0-15 -> N=lane, K = {0..7, 16..23}; lanes 16-31 -> N=lane-16, K = {8..15, 24..31}
__device__ __forceinline__ v16bf load_b_frag(const bf16_t* BT, int ldb, int col0, int k0, int lane) {
  const bf16_t* p = BT + (size_t)(col0 + (lane & 15)) * ldb + k0 + ((lane >> 4) << 3);
  union { uint4 u[2]; v16bf v; } t;
  t.u[0] = *(const uint4*)(p);        // K +0..7
  t.u[1] = *(const uint4*)(p + 16);   // K +16..23
  return t.v;
}

// 32x64 output tile per wave: 2 A-frags x 4 B-frags, f32 accum.
__device__ __forceinline__ void mma_tile_32x64(const bf16_t* A, const bf16_t* BT,
                                               int lda, int ldb, int kdim,
                                               int m0, int n0, int lane, v8f acc[2][4]) {
  v8f zero = {0.f,0.f,0.f,0.f,0.f,0.f,0.f,0.f};
  #pragma unroll
  for (int i = 0; i < 2; ++i)
    #pragma unroll
    for (int j = 0; j < 4; ++j) acc[i][j] = zero;

  for (int k0 = 0; k0 < kdim; k0 += 32) {
    v16bf a0 = load_a_frag(A, lda, m0,      k0, lane);
    v16bf a1 = load_a_frag(A, lda, m0 + 16, k0, lane);
    v16bf b0 = load_b_frag(BT, ldb, n0,      k0, lane);
    v16bf b1 = load_b_frag(BT, ldb, n0 + 16, k0, lane);
    v16bf b2 = load_b_frag(BT, ldb, n0 + 32, k0, lane);
    v16bf b3 = load_b_frag(BT, ldb, n0 + 48, k0, lane);
    acc[0][0] = __builtin_amdgcn_wmma_f32_16x16x32_bf16(false, a0, false, b0, (short)0, acc[0][0], false, false);
    acc[0][1] = __builtin_amdgcn_wmma_f32_16x16x32_bf16(false, a0, false, b1, (short)0, acc[0][1], false, false);
    acc[0][2] = __builtin_amdgcn_wmma_f32_16x16x32_bf16(false, a0, false, b2, (short)0, acc[0][2], false, false);
    acc[0][3] = __builtin_amdgcn_wmma_f32_16x16x32_bf16(false, a0, false, b3, (short)0, acc[0][3], false, false);
    acc[1][0] = __builtin_amdgcn_wmma_f32_16x16x32_bf16(false, a1, false, b0, (short)0, acc[1][0], false, false);
    acc[1][1] = __builtin_amdgcn_wmma_f32_16x16x32_bf16(false, a1, false, b1, (short)0, acc[1][1], false, false);
    acc[1][2] = __builtin_amdgcn_wmma_f32_16x16x32_bf16(false, a1, false, b2, (short)0, acc[1][2], false, false);
    acc[1][3] = __builtin_amdgcn_wmma_f32_16x16x32_bf16(false, a1, false, b3, (short)0, acc[1][3], false, false);
  }
}

// ---------- conversion kernels ----------

__global__ void cvt_bf16_kernel(const float* __restrict__ in, uint16_t* __restrict__ out, int n) {
  int i = blockIdx.x * blockDim.x + threadIdx.x;
  if (i < n) out[i] = f32_to_bf16_bits(in[i]);
}

__global__ void cvt_transpose_bf16_kernel(const float* __restrict__ in, uint16_t* __restrict__ out, int dim) {
  int i = blockIdx.x * blockDim.x + threadIdx.x;
  if (i < dim * dim) {
    int e = i / dim, d = i - e * dim;
    out[(size_t)e * dim + d] = f32_to_bf16_bits(in[(size_t)d * dim + e]);
  }
}

// ---------- GEMM 1: Q = E*W1, K = E*W2 (bf16 out) ----------
// grid: (D/256, N/64, B*2); block 256 (8 waves, 2x4 wave tiling -> 64x256 per WG)

__global__ void __launch_bounds__(256)
gemm_qk_kernel(const uint16_t* __restrict__ Ebf,
               const uint16_t* __restrict__ W1T,
               const uint16_t* __restrict__ W2T,
               uint16_t* __restrict__ Qbf,
               uint16_t* __restrict__ Kbf) {
  const int lane = threadIdx.x & 31;
  const int wave = threadIdx.x >> 5;
  const int b    = blockIdx.z >> 1;
  const int sel  = blockIdx.z & 1;

  const bf16_t* A  = (const bf16_t*)Ebf + (size_t)b * N_ * D_;
  const bf16_t* BT = (const bf16_t*)(sel ? W2T : W1T);
  uint16_t*    Out = (sel ? Kbf : Qbf) + (size_t)b * N_ * D_;

  const int m0 = blockIdx.y * 64 + (wave >> 2) * 32;
  const int n0 = blockIdx.x * 256 + (wave & 3) * 64;

  v8f acc[2][4];
  mma_tile_32x64(A, BT, D_, D_, D_, m0, n0, lane, acc);

  const int colbase = n0 + (lane & 15);
  const int rowadd  = (lane >> 4) * 8;
  #pragma unroll
  for (int i = 0; i < 2; ++i)
    #pragma unroll
    for (int j = 0; j < 4; ++j)
      #pragma unroll
      for (int r = 0; r < 8; ++r) {
        int row = m0 + 16 * i + rowadd + r;
        int col = colbase + 16 * j;
        Out[(size_t)row * D_ + col] = f32_to_bf16_bits(acc[i][j][r]);
      }
}

// ---------- GEMM 2: S = Q * K^T * scale (f32 out) ----------
// grid: (N/256, N/64, B); block 256

__global__ void __launch_bounds__(256)
gemm_scores_kernel(const uint16_t* __restrict__ Qbf,
                   const uint16_t* __restrict__ Kbf,
                   float* __restrict__ S, float scale) {
  const int lane = threadIdx.x & 31;
  const int wave = threadIdx.x >> 5;
  const int b    = blockIdx.z;

  const bf16_t* A  = (const bf16_t*)Qbf + (size_t)b * N_ * D_;
  const bf16_t* BT = (const bf16_t*)Kbf + (size_t)b * N_ * D_;  // K rows are K^T columns
  float* Out = S + (size_t)b * N_ * N_;

  const int m0 = blockIdx.y * 64 + (wave >> 2) * 32;
  const int n0 = blockIdx.x * 256 + (wave & 3) * 64;

  v8f acc[2][4];
  mma_tile_32x64(A, BT, D_, D_, D_, m0, n0, lane, acc);

  const int colbase = n0 + (lane & 15);
  const int rowadd  = (lane >> 4) * 8;
  #pragma unroll
  for (int i = 0; i < 2; ++i)
    #pragma unroll
    for (int j = 0; j < 4; ++j)
      #pragma unroll
      for (int r = 0; r < 8; ++r) {
        int row = m0 + 16 * i + rowadd + r;
        int col = colbase + 16 * j;
        Out[(size_t)row * N_ + col] = acc[i][j][r] * scale;
      }
}

// ---------- column softmax (axis = rows) ----------

__global__ void colsoftmax_stats_kernel(const float* __restrict__ S,
                                        float* __restrict__ cmax,
                                        float* __restrict__ csum) {
  int idx = blockIdx.x * blockDim.x + threadIdx.x;   // idx = b*N + m
  if (idx >= B_ * N_) return;
  int b = idx / N_;
  int m = idx - b * N_;
  const float* col = S + (size_t)b * N_ * N_ + m;
  float mx = -INFINITY, s = 0.0f;
  for (int n = 0; n < N_; ++n) {
    float x  = col[(size_t)n * N_];
    float nm = fmaxf(mx, x);
    s = s * __expf(mx - nm) + __expf(x - nm);
    mx = nm;
  }
  cmax[idx] = mx;
  csum[idx] = s;
}

__global__ void colsoftmax_norm_kernel(float* __restrict__ S,
                                       const float* __restrict__ cmax,
                                       const float* __restrict__ csum) {
  size_t i = (size_t)blockIdx.x * blockDim.x + threadIdx.x;
  size_t total = (size_t)B_ * N_ * N_;
  if (i >= total) return;
  size_t b = i / ((size_t)N_ * N_);
  int m = (int)(i % N_);
  size_t stat = b * N_ + m;
  S[i] = __expf(S[i] - cmax[stat]) / csum[stat];
}

// ---------- launcher ----------

extern "C" void kernel_launch(void* const* d_in, const int* in_sizes, int n_in,
                              void* d_out, int out_size, void* d_ws, size_t ws_size,
                              hipStream_t stream) {
  const float* E  = (const float*)d_in[0];
  const float* W1 = (const float*)d_in[1];
  const float* W2 = (const float*)d_in[2];
  float* S = (float*)d_out;

  char* w = (char*)d_ws;
  const size_t szEQK = (size_t)B_ * N_ * D_ * sizeof(uint16_t);   // 16 MB each
  const size_t szW   = (size_t)D_ * D_ * sizeof(uint16_t);        // 0.5 MB each
  uint16_t* Ebf = (uint16_t*)(w);
  uint16_t* Qbf = (uint16_t*)(w + szEQK);
  uint16_t* Kbf = (uint16_t*)(w + 2 * szEQK);
  uint16_t* W1T = (uint16_t*)(w + 3 * szEQK);
  uint16_t* W2T = (uint16_t*)(w + 3 * szEQK + szW);
  float*   cmax = (float*)  (w + 3 * szEQK + 2 * szW);
  float*   csum = (float*)  (w + 3 * szEQK + 2 * szW + (size_t)B_ * N_ * sizeof(float));

  // 1) convert inputs to bf16 (W transposed so B-fragments are contiguous)
  {
    int n = B_ * N_ * D_;
    cvt_bf16_kernel<<<(n + 255) / 256, 256, 0, stream>>>(E, Ebf, n);
    int nw = D_ * D_;
    cvt_transpose_bf16_kernel<<<(nw + 255) / 256, 256, 0, stream>>>(W1, W1T, D_);
    cvt_transpose_bf16_kernel<<<(nw + 255) / 256, 256, 0, stream>>>(W2, W2T, D_);
  }

  // 2) Q = E*W1, K = E*W2
  gemm_qk_kernel<<<dim3(D_ / 256, N_ / 64, B_ * 2), 256, 0, stream>>>(Ebf, W1T, W2T, Qbf, Kbf);

  // 3) S = Q*K^T / sqrt(D)
  const float scale = 0.044194173824159216f;  // 1/sqrt(512)
  gemm_scores_kernel<<<dim3(N_ / 256, N_ / 64, B_), 256, 0, stream>>>(Qbf, Kbf, S, scale);

  // 4) softmax along axis=1 (per-column over rows)
  colsoftmax_stats_kernel<<<(B_ * N_ + 255) / 256, 256, 0, stream>>>(S, cmax, csum);
  {
    size_t total = (size_t)B_ * N_ * N_;
    colsoftmax_norm_kernel<<<(unsigned)((total + 255) / 256), 256, 0, stream>>>(S, cmax, csum);
  }
}
